// MambaBlock_Channel_59923383714385
// MI455X (gfx1250) — compile-verified, matrix operand order checked by reading, not run on hardware
//
#include <hip/hip_runtime.h>
#include <hip/hip_bf16.h>

#define DI     1024
#define DSSZ   128
#define DMODEL 512
#define DTR    32
#define NBATCH 4
#define LSEQ   64
#define MROWS  (NBATCH*LSEQ)      // 256
#define DBC_N  (DTR + 2*DSSZ)     // 288

typedef _Float16 half_t;
typedef _Float16 v16h __attribute__((ext_vector_type(16)));
typedef float    v8f  __attribute__((ext_vector_type(8)));

// ---------------------------------------------------------------------------
// 1) Fused adaptive avg+max pool: [4,512,64,64] -> xG,xM [B, L=64, 512] f32.
//    Single pass over the 33.5 MB of input (the roofline-dominant traffic).
// ---------------------------------------------------------------------------
__global__ void pool_kernel(const float* __restrict__ x1, const float* __restrict__ x2,
                            float* __restrict__ xG, float* __restrict__ xM)
{
    int bc = blockIdx.x;                 // 0..2047
    int b  = bc >> 9;
    int c  = bc & 511;
    const float* plane = (c < 256)
        ? (x1 + (size_t)(b * 256 + c) * 4096)
        : (x2 + (size_t)(b * 256 + (c - 256)) * 4096);
    int t  = threadIdx.x;                // cell 0..63
    int cy = t >> 3, cx = t & 7;
    float s = 0.f, mx = -3.402823466e38f;
    #pragma unroll
    for (int ky = 0; ky < 8; ++ky) {
        const float* row = plane + (cy * 8 + ky) * 64 + cx * 8;
        #pragma unroll
        for (int kx = 0; kx < 8; ++kx) {
            float v = row[kx];
            s += v;
            mx = fmaxf(mx, v);
        }
    }
    size_t o = (size_t)(b * LSEQ + t) * DMODEL + c;
    xG[o] = s * (1.f / 64.f);
    xM[o] = mx;
}

// ---------------------------------------------------------------------------
// 2) f32 -> f16 weight conversion
// ---------------------------------------------------------------------------
__global__ void cvt_kernel(const float* __restrict__ src, half_t* __restrict__ dst, int n)
{
    int i = blockIdx.x * blockDim.x + threadIdx.x;
    if (i < n) dst[i] = (half_t)src[i];
}

// ---------------------------------------------------------------------------
// 3) WMMA GEMM: C[M,N] = A[M,K](f32, row stride lda) * W[N,K]^T (f16).
//    One wave computes a 16x(16*NT) strip.  NT must divide (N/16) exactly,
//    so the body is straight-line: A fragment converted once per k-step, NT
//    B-fragments loaded in one clause, NT independent WMMAs back-to-back.
//    Wave id is readfirstlane'd -> scalar control flow, EXEC stays all-1s.
//    Fragment packing per ISA 7.12.2: lane half h owns K = 8h..8h+7 and
//    16+8h..+7 (contiguous 8-chunks -> b128 loads).
//    ACT: 0=none, 1=softplus.  BIAS: add bias[col].
// ---------------------------------------------------------------------------
template<int ACT, bool BIAS, int NT>
__global__ void gemm_wmma_kernel(const float* __restrict__ A, int lda,
                                 const half_t* __restrict__ W,
                                 const float* __restrict__ bias,
                                 float* __restrict__ C,
                                 int M, int N, int K)
{
    int wib   = __builtin_amdgcn_readfirstlane((int)threadIdx.x >> 5);   // scalar wave id
    int gwave = (int)blockIdx.x * ((int)blockDim.x >> 5) + wib;
    int lane  = (int)threadIdx.x & 31;
    int tilesN  = N >> 4;
    int tilesM  = M >> 4;
    int groupsN = tilesN / NT;                 // exact by construction
    if (gwave >= tilesM * groupsN) return;     // scalar branch (normally no-op)
    int tm = gwave / groupsN;
    int tg = gwave % groupsN;
    int h = lane >> 4;                          // lane half
    int r = lane & 15;                          // A row / W row (=B col)

    const float*  Arow = A + (size_t)(tm * 16 + r) * lda;
    const half_t* Wrow[NT];
    #pragma unroll
    for (int j = 0; j < NT; ++j)
        Wrow[j] = W + (size_t)((tg * NT + j) * 16 + r) * K;

    v8f acc[NT] = {};
    for (int k0 = 0; k0 < K; k0 += 32) {
        if (k0 + 32 < K) {
            __builtin_prefetch(Arow + k0 + 32, 0, 1);        // global_prefetch_b8
            __builtin_prefetch(Wrow[0] + k0 + 32, 0, 1);
        }
        v16h a;
        #pragma unroll
        for (int i = 0; i < 8; ++i) a[i]     = (half_t)Arow[k0 + 8 * h + i];
        #pragma unroll
        for (int i = 0; i < 8; ++i) a[8 + i] = (half_t)Arow[k0 + 16 + 8 * h + i];

        v16h bm[NT];
        #pragma unroll
        for (int j = 0; j < NT; ++j) {
            #pragma unroll
            for (int i = 0; i < 8; ++i) bm[j][i]     = Wrow[j][k0 + 8 * h + i];
            #pragma unroll
            for (int i = 0; i < 8; ++i) bm[j][8 + i] = Wrow[j][k0 + 16 + 8 * h + i];
        }
        #pragma unroll
        for (int j = 0; j < NT; ++j)
            acc[j] = __builtin_amdgcn_wmma_f32_16x16x32_f16(false, a, false, bm[j],
                                                            (short)0, acc[j], false, false);
    }

    // C layout: VGPR i -> M = tm*16 + i + 8*h, N = tn*16 + (lane&15)
    #pragma unroll
    for (int j = 0; j < NT; ++j) {
        int col = (tg * NT + j) * 16 + r;
        float bv = BIAS ? bias[col] : 0.f;
        #pragma unroll
        for (int i = 0; i < 8; ++i) {
            int row = tm * 16 + i + 8 * h;
            float v = acc[j][i] + bv;
            if (ACT == 1) v = (v > 20.f) ? v : log1pf(__expf(v));   // softplus
            C[(size_t)row * N + col] = v;
        }
    }
}

// ---------------------------------------------------------------------------
// 4) Causal depthwise conv (k=4) over L, + SiLU.  xz: [B*L, 2*DI], x half.
// ---------------------------------------------------------------------------
__global__ void conv_silu_kernel(const float* __restrict__ xz,
                                 const float* __restrict__ convw,
                                 const float* __restrict__ convb,
                                 float* __restrict__ xc)
{
    int idx = blockIdx.x * blockDim.x + threadIdx.x;   // over MROWS*DI
    if (idx >= MROWS * DI) return;
    int e  = idx & (DI - 1);
    int bl = idx >> 10;
    int l  = bl & (LSEQ - 1);
    int b  = bl >> 6;
    float acc = convb[e];
    #pragma unroll
    for (int k = 0; k < 4; ++k) {
        int ls = l - 3 + k;
        if (ls >= 0)
            acc = fmaf(xz[(size_t)(b * LSEQ + ls) * (2 * DI) + e], convw[e * 4 + k], acc);
    }
    xc[idx] = acc / (1.f + __expf(-acc));              // silu
}

// ---------------------------------------------------------------------------
// 5) Selective scan: one wave per (b,e), 4 states/lane (S=128), L=64 steps.
//    h = exp(delta*A)*h + delta*B*x ; y = h·C via shfl_xor tree.
//    Epilogue fused: y = (scan + x*D) * silu(z).
// ---------------------------------------------------------------------------
__global__ void scan_kernel(const float* __restrict__ dbc,    // [B*L, 288]
                            const float* __restrict__ delta,  // [B*L, DI]
                            const float* __restrict__ xc,     // [B*L, DI]
                            const float* __restrict__ xz,     // [B*L, 2*DI] (z at +DI)
                            const float* __restrict__ Alog,   // [DI, S]
                            const float* __restrict__ Dp,     // [DI]
                            float* __restrict__ y)            // [B*L, DI]
{
    int wid  = (int)((blockIdx.x * (size_t)blockDim.x + threadIdx.x) >> 5);
    int lane = (int)threadIdx.x & 31;
    if (wid >= NBATCH * DI) return;
    int e = wid & (DI - 1);
    int b = wid >> 10;

    float Aes[4], h[4] = {0.f, 0.f, 0.f, 0.f};
    #pragma unroll
    for (int j = 0; j < 4; ++j)
        Aes[j] = -__expf(Alog[(size_t)e * DSSZ + lane + 32 * j]);
    float Dv = Dp[e];

    for (int l = 0; l < LSEQ; ++l) {
        int bl = b * LSEQ + l;
        float dlt = delta[(size_t)bl * DI + e];
        float xv  = xc[(size_t)bl * DI + e];
        const float* bc = dbc + (size_t)bl * DBC_N;
        float part = 0.f;
        #pragma unroll
        for (int j = 0; j < 4; ++j) {
            int s = lane + 32 * j;
            float dA = __expf(dlt * Aes[j]);
            h[j] = fmaf(dA, h[j], dlt * bc[DTR + s] * xv);
            part = fmaf(h[j], bc[DTR + DSSZ + s], part);
        }
        #pragma unroll
        for (int off = 16; off > 0; off >>= 1)
            part += __shfl_xor(part, off, 32);
        if (lane == 0) {
            float zv = xz[(size_t)bl * (2 * DI) + DI + e];
            float silz = zv / (1.f + __expf(-zv));
            y[(size_t)bl * DI + e] = (part + xv * Dv) * silz;
        }
    }
}

// ---------------------------------------------------------------------------
// 6) out = sigmoid(gG + gM)
// ---------------------------------------------------------------------------
__global__ void add_sigmoid_kernel(const float* __restrict__ a, const float* __restrict__ b,
                                   float* __restrict__ out, int n)
{
    int i = blockIdx.x * blockDim.x + threadIdx.x;
    if (i < n) {
        float v = a[i] + b[i];
        out[i] = 1.f / (1.f + __expf(-v));
    }
}

// ---------------------------------------------------------------------------
// Host orchestration
// ---------------------------------------------------------------------------
extern "C" void kernel_launch(void* const* d_in, const int* in_sizes, int n_in,
                              void* d_out, int out_size, void* d_ws, size_t ws_size,
                              hipStream_t stream)
{
    (void)in_sizes; (void)n_in; (void)out_size; (void)ws_size;
    const float* x1 = (const float*)d_in[0];
    const float* x2 = (const float*)d_in[1];
    struct MP { const float *Win, *convw, *convb, *Wx, *Wdt, *bdt, *Alog, *D, *Wout; };
    MP mp[2];
    for (int m = 0; m < 2; ++m) {
        int k = 2 + m * 9;
        mp[m].Win   = (const float*)d_in[k + 0];
        mp[m].convw = (const float*)d_in[k + 1];
        mp[m].convb = (const float*)d_in[k + 2];
        mp[m].Wx    = (const float*)d_in[k + 3];
        mp[m].Wdt   = (const float*)d_in[k + 4];
        mp[m].bdt   = (const float*)d_in[k + 5];
        mp[m].Alog  = (const float*)d_in[k + 6];
        mp[m].D     = (const float*)d_in[k + 7];
        mp[m].Wout  = (const float*)d_in[k + 8];
    }
    const float* lW[2] = { (const float*)d_in[20], (const float*)d_in[22] };
    const float* lb[2] = { (const float*)d_in[21], (const float*)d_in[23] };

    // deterministic bump allocator over d_ws
    char* base = (char*)d_ws;
    size_t off = 0;
    auto alloc = [&](size_t bytes) -> void* {
        void* p = base + off;
        off = (off + bytes + 255) & ~(size_t)255;
        return p;
    };

    float* xPool[2];
    xPool[0] = (float*)alloc((size_t)MROWS * DMODEL * 4);   // xG
    xPool[1] = (float*)alloc((size_t)MROWS * DMODEL * 4);   // xM
    half_t *Win16[2], *Wx16[2], *Wdt16[2], *Wout16[2], *lW16[2];
    float  *xzb[2], *xcb[2], *dbcb[2], *deltab[2], *yb[2], *yob[2], *gb[2];
    for (int m = 0; m < 2; ++m) {
        Win16[m]  = (half_t*)alloc((size_t)2 * DI * DMODEL * 2);
        Wx16[m]   = (half_t*)alloc((size_t)DBC_N * DI * 2);
        Wdt16[m]  = (half_t*)alloc((size_t)DI * DTR * 2);
        Wout16[m] = (half_t*)alloc((size_t)DMODEL * DI * 2);
        lW16[m]   = (half_t*)alloc((size_t)(DMODEL / 2) * DMODEL * 2);
        xzb[m]    = (float*)alloc((size_t)MROWS * 2 * DI * 4);
        xcb[m]    = (float*)alloc((size_t)MROWS * DI * 4);
        dbcb[m]   = (float*)alloc((size_t)MROWS * DBC_N * 4);
        deltab[m] = (float*)alloc((size_t)MROWS * DI * 4);
        yb[m]     = (float*)alloc((size_t)MROWS * DI * 4);
        yob[m]    = (float*)alloc((size_t)MROWS * DMODEL * 4);
        gb[m]     = (float*)alloc((size_t)MROWS * (DMODEL / 2) * 4);
    }

    // --- pooling (single pass over x1/x2) ---
    pool_kernel<<<NBATCH * DMODEL, 64, 0, stream>>>(x1, x2, xPool[0], xPool[1]);

    // --- weight conversions ---
    auto cvt = [&](const float* s, half_t* d, int n) {
        cvt_kernel<<<(n + 255) / 256, 256, 0, stream>>>(s, d, n);
    };
    for (int m = 0; m < 2; ++m) {
        cvt(mp[m].Win,  Win16[m],  2 * DI * DMODEL);
        cvt(mp[m].Wx,   Wx16[m],   DBC_N * DI);
        cvt(mp[m].Wdt,  Wdt16[m],  DI * DTR);
        cvt(mp[m].Wout, Wout16[m], DMODEL * DI);
        cvt(lW[m],      lW16[m],   (DMODEL / 2) * DMODEL);
    }

    // waves = tilesM * (tilesN/NT); NT chosen so it divides exactly.
    auto gemm_blocks = [](int M, int N, int NT) {
        int waves = (M >> 4) * ((N >> 4) / NT);
        return (waves + 7) / 8;
    };

    for (int m = 0; m < 2; ++m) {
        // xz = x @ Win^T : [256,2048]  (128 N-tiles, NT=4)
        gemm_wmma_kernel<0, false, 4><<<gemm_blocks(MROWS, 2 * DI, 4), 256, 0, stream>>>(
            xPool[m], DMODEL, Win16[m], nullptr, xzb[m], MROWS, 2 * DI, DMODEL);
        // causal depthwise conv + silu
        conv_silu_kernel<<<(MROWS * DI + 255) / 256, 256, 0, stream>>>(
            xzb[m], mp[m].convw, mp[m].convb, xcb[m]);
        // dbc = xc @ Wx^T : [256,288]  (18 N-tiles, NT=2)
        gemm_wmma_kernel<0, false, 2><<<gemm_blocks(MROWS, DBC_N, 2), 256, 0, stream>>>(
            xcb[m], DI, Wx16[m], nullptr, dbcb[m], MROWS, DBC_N, DI);
        // delta = softplus(dt @ Wdt^T + bdt) : A = dbc[:, :32] with lda=288  (64 N-tiles, NT=4)
        gemm_wmma_kernel<1, true, 4><<<gemm_blocks(MROWS, DI, 4), 256, 0, stream>>>(
            dbcb[m], DBC_N, Wdt16[m], mp[m].bdt, deltab[m], MROWS, DI, DTR);
        // selective scan (fused +x*D, *silu(z))
        scan_kernel<<<(NBATCH * DI) / 4, 128, 0, stream>>>(
            dbcb[m], deltab[m], xcb[m], xzb[m], mp[m].Alog, mp[m].D, yb[m]);
        // yo = y @ Wout^T : [256,512]  (32 N-tiles, NT=4)
        gemm_wmma_kernel<0, false, 4><<<gemm_blocks(MROWS, DMODEL, 4), 256, 0, stream>>>(
            yb[m], DI, Wout16[m], nullptr, yob[m], MROWS, DMODEL, DI);
        // g = yo @ lW^T + lb : [256,256]  (16 N-tiles, NT=4)
        gemm_wmma_kernel<0, true, 4><<<gemm_blocks(MROWS, DMODEL / 2, 4), 256, 0, stream>>>(
            yob[m], DMODEL, lW16[m], lb[m], gb[m], MROWS, DMODEL / 2, DMODEL);
    }

    int n_out = MROWS * (DMODEL / 2);
    add_sigmoid_kernel<<<(n_out + 255) / 256, 256, 0, stream>>>(
        gb[0], gb[1], (float*)d_out, n_out);
}
